// LinearTransformerBlock_6949257085174
// MI455X (gfx1250) — compile-verified, matrix operand order checked by reading, not run on hardware
//
#include <hip/hip_runtime.h>
#include <math.h>

#define D_MODEL 1024
#define SEQ     4096
#define NB      4
#define NH      16
#define DH      64
#define FFN     4096
#define N_TOK   (NB * SEQ)   // 16384

typedef __attribute__((ext_vector_type(16))) __bf16 v16bf;
typedef __attribute__((ext_vector_type(8)))  __bf16 v8bf;
typedef __attribute__((ext_vector_type(8)))  float  v8f;

__device__ __forceinline__ unsigned short f2bf(float f) {
    union { float f; unsigned u; } v; v.f = f;
    unsigned u = v.u + 0x7FFFu + ((v.u >> 16) & 1u);   // round-to-nearest-even
    return (unsigned short)(u >> 16);
}
__device__ __forceinline__ float bf2f(unsigned short s) {
    union { unsigned u; float f; } v; v.u = ((unsigned)s) << 16; return v.f;
}
__device__ __forceinline__ v8f v8f_zero() {
    v8f z;
    #pragma unroll
    for (int i = 0; i < 8; ++i) z[i] = 0.f;
    return z;
}

// Async DMA of one 16-byte chunk: global -> LDS, tracked by ASYNCcnt.
// VDST VGPR carries the per-lane LDS byte address (flat addr low 32 bits),
// VADDR the 64-bit global address (GV mode, SADDR=off).
__device__ __forceinline__ void async_g2l_b128(const void* lds_dst, const void* gsrc) {
    unsigned lds_off = (unsigned)(unsigned long long)(uintptr_t)lds_dst;
    unsigned long long ga = (unsigned long long)(uintptr_t)gsrc;
    asm volatile("global_load_async_to_lds_b128 %0, %1, off"
                 :: "v"(lds_off), "v"(ga) : "memory");
}
__device__ __forceinline__ void wait_asynccnt0() {
    asm volatile("s_wait_asynccnt 0x0" ::: "memory");
}

// Load one 16x32 bf16 operand fragment from an LDS tile stored row-major with
// `stride` (in bf16 elems).  Per ISA 7.12.2 (16-bit A-matrix 16x32):
//   lanes 0-15 : row = lane,     K chunks [0..7] and [16..23]
//   lanes 16-31: row = lane-16,  K chunks [8..15] and [24..31]
// B uses the same per-lane layout when stored N-major/K-contiguous (B^T).
__device__ __forceinline__ v16bf lds_frag(const unsigned short* tileBase, int stride, int lane) {
    int row  = lane & 15;
    int koff = (lane >> 4) << 3;            // 0 or 8 elems
    const unsigned short* p = tileBase + row * stride + koff;
    v8bf lo = *(const v8bf*)(p);            // ds_load_b128
    v8bf hi = *(const v8bf*)(p + 16);       // ds_load_b128
    return __builtin_shufflevector(lo, hi, 0,1,2,3,4,5,6,7,8,9,10,11,12,13,14,15);
}

__device__ __forceinline__ v8f wmma_bf16(v16bf a, v16bf b, v8f c) {
    return __builtin_amdgcn_wmma_f32_16x16x32_bf16(false, a, false, b, (short)0, c, false, false);
}

// ---------------------------------------------------------------------------
// fp32 -> bf16 bulk convert (4 elems / thread)
// ---------------------------------------------------------------------------
__global__ __launch_bounds__(256) void cvt_f32_bf16_kernel(const float* __restrict__ in,
                                                           unsigned short* __restrict__ out,
                                                           int nvec) {
    int i = blockIdx.x * 256 + threadIdx.x;
    if (i >= nvec) return;
    float4 f = ((const float4*)in)[i];
    union { unsigned short s[4]; uint2 u; } r;
    r.s[0] = f2bf(f.x); r.s[1] = f2bf(f.y); r.s[2] = f2bf(f.z); r.s[3] = f2bf(f.w);
    ((uint2*)out)[i] = r.u;
}

// ---------------------------------------------------------------------------
// Transpose + convert: in fp32 [R][C]  ->  out bf16 [C][R]
// ---------------------------------------------------------------------------
__global__ __launch_bounds__(256) void transpose_cvt_kernel(const float* __restrict__ in,
                                                            unsigned short* __restrict__ out,
                                                            int R, int C) {
    __shared__ float tile[32][33];
    int tx = threadIdx.x & 31;
    int ty = threadIdx.x >> 5;              // 0..7
    int c  = blockIdx.x * 32 + tx;
    #pragma unroll
    for (int i = 0; i < 32; i += 8) {
        int r = blockIdx.y * 32 + ty + i;
        tile[ty + i][tx] = in[(size_t)r * C + c];
    }
    __syncthreads();
    int rr = blockIdx.y * 32 + tx;          // output col (= input row)
    #pragma unroll
    for (int i = 0; i < 32; i += 8) {
        int cc = blockIdx.x * 32 + ty + i;  // output row (= input col)
        out[(size_t)cc * R + rr] = f2bf(tile[tx][ty + i]);
    }
}

// ---------------------------------------------------------------------------
// Main bf16 WMMA GEMM:  C[M][N] = A[M][K] * Bt[N][K]^T  (+ fused epilogue)
// Block: 256 thr = 8 waves; block tile 128x128; wave tile 64x32; BK = 32.
// Double-buffered LDS, staged with GLOBAL_LOAD_ASYNC_TO_LDS_B128 (ASYNCcnt).
// ---------------------------------------------------------------------------
enum { EPI_ELU1 = 0, EPI_BIAS = 1, EPI_RELU = 2, EPI_RESID = 3 };

template <int MODE>
__global__ __launch_bounds__(256) void gemm_bf16_kernel(
        const unsigned short* __restrict__ A,
        const unsigned short* __restrict__ Bt,
        int M, int N, int K,
        const float* __restrict__ bias,
        const float* __restrict__ resid,        // fp32 [M][N], MODE==EPI_RESID
        unsigned short* __restrict__ outBf,     // bf16 [M][N], MODE!=EPI_RESID
        float* __restrict__ outF) {             // fp32 [M][N], MODE==EPI_RESID
    constexpr int BM = 128, BN = 128, BK = 32, LS = 40;   // 40-elem stride: conflict-free b128
    __shared__ unsigned short ldsA[2][BM * LS];
    __shared__ unsigned short ldsB[2][BN * LS];

    const int t     = threadIdx.x;
    const int lane  = t & 31;
    const int wave  = t >> 5;
    const int waveM = wave >> 2;        // 0..1
    const int waveN = wave & 3;         // 0..3
    const int m0    = blockIdx.y * BM;
    const int n0    = blockIdx.x * BN;
    const int lr    = t >> 2;           // 0..63
    const int lc    = (t & 3) << 3;     // 0,8,16,24

    // async-stage one BK tile (A and B) into LDS buffer `buf`
    auto stage = [&](int k0, int buf) {
        const unsigned short* gA = A  + (size_t)(m0 + lr) * K + k0 + lc;
        const unsigned short* gB = Bt + (size_t)(n0 + lr) * K + k0 + lc;
        async_g2l_b128(&ldsA[buf][lr * LS + lc],        gA);
        async_g2l_b128(&ldsA[buf][(lr + 64) * LS + lc], gA + (size_t)64 * K);
        async_g2l_b128(&ldsB[buf][lr * LS + lc],        gB);
        async_g2l_b128(&ldsB[buf][(lr + 64) * LS + lc], gB + (size_t)64 * K);
    };

    v8f acc[4][2];
    #pragma unroll
    for (int mi = 0; mi < 4; ++mi)
        #pragma unroll
        for (int ni = 0; ni < 2; ++ni) acc[mi][ni] = v8f_zero();

    const int nk = K / BK;
    stage(0, 0);
    wait_asynccnt0();
    __syncthreads();

    for (int kt = 0; kt < nk; ++kt) {
        const int cur = kt & 1;
        if (kt + 1 < nk) stage((kt + 1) * BK, cur ^ 1);          // DMA overlaps compute
        if (kt + 2 < nk) {                                       // warm L2 one tile further
            __builtin_prefetch(A  + (size_t)(m0 + lr) * K + (kt + 2) * BK + lc, 0, 0);
            __builtin_prefetch(Bt + (size_t)(n0 + lr) * K + (kt + 2) * BK + lc, 0, 0);
        }

        v16bf afr[4];
        #pragma unroll
        for (int mi = 0; mi < 4; ++mi)
            afr[mi] = lds_frag(&ldsA[cur][(waveM * 64 + mi * 16) * LS], LS, lane);
        #pragma unroll
        for (int ni = 0; ni < 2; ++ni) {
            v16bf bfr = lds_frag(&ldsB[cur][(waveN * 32 + ni * 16) * LS], LS, lane);
            #pragma unroll
            for (int mi = 0; mi < 4; ++mi)
                acc[mi][ni] = wmma_bf16(afr[mi], bfr, acc[mi][ni]);
        }

        wait_asynccnt0();        // own async stores to LDS landed (next buffer)
        __syncthreads();         // + everyone done reading `cur` before restage
    }

    // Epilogue.  C/D layout: VGPR i -> (M = i or i+8 for upper lanes, N = lane&15)
    const int lane15 = lane & 15;
    const int rsel   = (lane >> 4) << 3;
    #pragma unroll
    for (int mi = 0; mi < 4; ++mi) {
        #pragma unroll
        for (int ni = 0; ni < 2; ++ni) {
            int rBase = m0 + waveM * 64 + mi * 16 + rsel;
            int c     = n0 + waveN * 32 + ni * 16 + lane15;
            float bcol = bias[c];
            #pragma unroll
            for (int i = 0; i < 8; ++i) {
                int r = rBase + i;
                float v = acc[mi][ni][i] + bcol;
                if (MODE == EPI_ELU1) v = (v > 0.f) ? (v + 1.f) : __expf(v);  // elu(x)+1
                if (MODE == EPI_RELU) v = fmaxf(v, 0.f);
                size_t idx = (size_t)r * N + c;
                if (MODE == EPI_RESID) {
                    v += resid[idx];
                    outF[idx] = v;
                } else {
                    outBf[idx] = f2bf(v);
                }
            }
        }
    }
}

// ---------------------------------------------------------------------------
// KV state + Ksum:  per (n,h):  KV[m][d] = sum_s V[s][m]*K[s][d],  Ksum[d]=sum_s K[s][d]
// ~1% of total FLOPs -> VALU reduction with LDS staging.
// ---------------------------------------------------------------------------
__global__ __launch_bounds__(256) void kv_state_kernel(
        const unsigned short* __restrict__ Kf, const unsigned short* __restrict__ Vb,
        unsigned short* __restrict__ KVb, float* __restrict__ Ksum) {
    __shared__ unsigned short Ks[32 * 64];
    __shared__ unsigned short Vs[32 * 64];
    const int nh = blockIdx.x;
    const int n  = nh >> 4, h = nh & 15;
    const int t  = threadIdx.x;
    const int tm = t >> 4, td = t & 15;         // thread owns 4x4 block (m,d)
    const unsigned short* Kp = Kf + ((size_t)n * SEQ) * D_MODEL + h * DH;
    const unsigned short* Vp = Vb + ((size_t)n * SEQ) * D_MODEL + h * DH;

    float acc[4][4];
    #pragma unroll
    for (int a = 0; a < 4; ++a)
        #pragma unroll
        for (int b = 0; b < 4; ++b) acc[a][b] = 0.f;
    float ks4[4] = {0.f, 0.f, 0.f, 0.f};

    const int lrr = t >> 3, lcc = (t & 7) << 3;
    for (int s0 = 0; s0 < SEQ; s0 += 32) {
        __syncthreads();
        *(uint4*)(Ks + lrr * 64 + lcc) = *(const uint4*)(Kp + (size_t)(s0 + lrr) * D_MODEL + lcc);
        *(uint4*)(Vs + lrr * 64 + lcc) = *(const uint4*)(Vp + (size_t)(s0 + lrr) * D_MODEL + lcc);
        __syncthreads();
        #pragma unroll 4
        for (int s = 0; s < 32; ++s) {
            float kd[4], vm[4];
            #pragma unroll
            for (int b = 0; b < 4; ++b) kd[b] = bf2f(Ks[s * 64 + td * 4 + b]);
            #pragma unroll
            for (int a = 0; a < 4; ++a) vm[a] = bf2f(Vs[s * 64 + tm * 4 + a]);
            #pragma unroll
            for (int a = 0; a < 4; ++a)
                #pragma unroll
                for (int b = 0; b < 4; ++b) acc[a][b] += vm[a] * kd[b];
            if (tm == 0) {
                #pragma unroll
                for (int b = 0; b < 4; ++b) ks4[b] += kd[b];
            }
        }
    }
    unsigned short* kvp = KVb + (size_t)nh * DH * DH;   // [m][d], d contiguous
    #pragma unroll
    for (int a = 0; a < 4; ++a)
        #pragma unroll
        for (int b = 0; b < 4; ++b)
            kvp[(tm * 4 + a) * DH + td * 4 + b] = f2bf(acc[a][b]);
    if (tm == 0) {
        #pragma unroll
        for (int b = 0; b < 4; ++b) Ksum[(size_t)nh * DH + td * 4 + b] = ks4[b];
    }
}

// ---------------------------------------------------------------------------
// Z:  z[n,l,h] = 1 / (sum_d Q[n,l,h,d] * Ksum[n,h,d] + 1e-6)
// ---------------------------------------------------------------------------
__global__ __launch_bounds__(256) void z_kernel(const unsigned short* __restrict__ Qf,
                                                const float* __restrict__ Ksum,
                                                float* __restrict__ z) {
    __shared__ float red[256];
    const int tok = blockIdx.x;              // n*SEQ + l
    const int n   = tok >> 12;
    const int t   = threadIdx.x;
    const int h   = t >> 4, j = t & 15;
    const unsigned short* q = Qf + (size_t)tok * D_MODEL + h * DH + j * 4;
    const float* ks = Ksum + ((size_t)(n * NH + h)) * DH + j * 4;
    float p = 0.f;
    #pragma unroll
    for (int i = 0; i < 4; ++i) p += bf2f(q[i]) * ks[i];
    red[t] = p;
    __syncthreads();
    if (t < NH) {
        float s = 0.f;
        #pragma unroll
        for (int j2 = 0; j2 < 16; ++j2) s += red[t * 16 + j2];
        z[(size_t)tok * NH + t] = 1.f / (s + 1e-6f);
    }
}

// ---------------------------------------------------------------------------
// Per-head attention output:  out[l][m] = z[l] * sum_d Q[l][d] * KV[m][d]
// WMMA GEMM, M=4096 (per n), N=64 (head), K=64.  Block = 128 rows of one head.
// ---------------------------------------------------------------------------
__global__ __launch_bounds__(256) void attn_out_kernel(
        const unsigned short* __restrict__ Qf,
        const unsigned short* __restrict__ KVb,
        const float* __restrict__ z,
        unsigned short* __restrict__ attn) {
    constexpr int LS = 72;                   // 64 + 16B pad, conflict-free b128
    __shared__ unsigned short ldsQ[128 * LS];
    __shared__ unsigned short ldsKV[64 * LS];
    const int t    = threadIdx.x;
    const int lane = t & 31;
    const int wave = t >> 5;                 // 8 waves -> 16 rows each
    const int l0   = blockIdx.x * 128;
    const int h    = blockIdx.y;
    const int n    = blockIdx.z;

    const unsigned short* Qp = Qf  + ((size_t)n * SEQ) * D_MODEL + h * DH;
    const unsigned short* Kp = KVb + ((size_t)(n * NH + h)) * DH * DH;

    #pragma unroll
    for (int i = 0; i < 4; ++i) {            // Q tile 128x64: 1024 16B chunks
        int id = t + i * 256;
        int r = id >> 3, c = (id & 7) << 3;
        *(v8bf*)(ldsQ + r * LS + c) = *(const v8bf*)(Qp + (size_t)(l0 + r) * D_MODEL + c);
    }
    #pragma unroll
    for (int i = 0; i < 2; ++i) {            // KV tile 64x64: 512 16B chunks
        int id = t + i * 256;
        int r = id >> 3, c = (id & 7) << 3;
        *(v8bf*)(ldsKV + r * LS + c) = *(const v8bf*)(Kp + r * DH + c);
    }
    __syncthreads();

    v8f acc[4];
    #pragma unroll
    for (int ni = 0; ni < 4; ++ni) acc[ni] = v8f_zero();
    #pragma unroll
    for (int k0 = 0; k0 < DH; k0 += 32) {
        v16bf afr = lds_frag(ldsQ + (wave * 16) * LS + k0, LS, lane);
        #pragma unroll
        for (int ni = 0; ni < 4; ++ni) {
            v16bf bfr = lds_frag(ldsKV + (ni * 16) * LS + k0, LS, lane);
            acc[ni] = wmma_bf16(afr, bfr, acc[ni]);
        }
    }

    const int lane15 = lane & 15;
    const int rsel   = (lane >> 4) << 3;
    #pragma unroll
    for (int ni = 0; ni < 4; ++ni) {
        int c = ni * 16 + lane15;
        #pragma unroll
        for (int i = 0; i < 8; ++i) {
            int l = l0 + wave * 16 + rsel + i;
            float zz = z[((size_t)n * SEQ + l) * NH + h];
            attn[((size_t)n * SEQ + l) * D_MODEL + h * DH + c] = f2bf(acc[ni][i] * zz);
        }
    }
}

// ---------------------------------------------------------------------------
// LayerNorm over D_MODEL=1024 (4 elems/thread); may run in place; optional bf16 copy.
// ---------------------------------------------------------------------------
__global__ __launch_bounds__(256) void layernorm_kernel(const float* __restrict__ in,
                                                        const float* __restrict__ gamma,
                                                        const float* __restrict__ beta,
                                                        float* __restrict__ outF,
                                                        unsigned short* __restrict__ outBf) {
    __shared__ float r1[256], r2[256];
    const int row = blockIdx.x;
    const int t   = threadIdx.x;
    const float* p = in + (size_t)row * D_MODEL;
    float v[4], s = 0.f, ss = 0.f;
    #pragma unroll
    for (int i = 0; i < 4; ++i) { v[i] = p[t + i * 256]; s += v[i]; ss += v[i] * v[i]; }
    r1[t] = s; r2[t] = ss;
    __syncthreads();
    for (int o = 128; o > 0; o >>= 1) {
        if (t < o) { r1[t] += r1[t + o]; r2[t] += r2[t + o]; }
        __syncthreads();
    }
    float mu   = r1[0] * (1.f / 1024.f);
    float var  = r2[0] * (1.f / 1024.f) - mu * mu;
    float rstd = rsqrtf(var + 1e-5f);
    #pragma unroll
    for (int i = 0; i < 4; ++i) {
        int c = t + i * 256;
        float o = (v[i] - mu) * rstd * gamma[c] + beta[c];
        outF[(size_t)row * D_MODEL + c] = o;
        if (outBf) outBf[(size_t)row * D_MODEL + c] = f2bf(o);
    }
}

// ---------------------------------------------------------------------------
extern "C" void kernel_launch(void* const* d_in, const int* in_sizes, int n_in,
                              void* d_out, int out_size, void* d_ws, size_t ws_size,
                              hipStream_t stream) {
    const float* x      = (const float*)d_in[0];
    const float* Wq     = (const float*)d_in[1];
    const float* bq     = (const float*)d_in[2];
    const float* Wk     = (const float*)d_in[3];
    const float* bk     = (const float*)d_in[4];
    const float* Wv     = (const float*)d_in[5];
    const float* bv     = (const float*)d_in[6];
    const float* Wo     = (const float*)d_in[7];
    const float* bo     = (const float*)d_in[8];
    const float* W1     = (const float*)d_in[9];
    const float* b1     = (const float*)d_in[10];
    const float* W2     = (const float*)d_in[11];
    const float* b2     = (const float*)d_in[12];
    const float* gamma1 = (const float*)d_in[13];
    const float* beta1  = (const float*)d_in[14];
    const float* gamma2 = (const float*)d_in[15];
    const float* beta2  = (const float*)d_in[16];
    float* outF = (float*)d_out;

    char* ws = (char*)d_ws;
    const size_t MB = 1ull << 20;
    // Region A (0..128MB): Qf|Kf|Vb|attn (4x32MB), later reused as ffn1 (128MB bf16)
    unsigned short* Qf   = (unsigned short*)(ws + 0 * MB);
    unsigned short* Kf   = (unsigned short*)(ws + 32 * MB);
    unsigned short* Vb   = (unsigned short*)(ws + 64 * MB);
    unsigned short* attn = (unsigned short*)(ws + 96 * MB);
    unsigned short* ffn1 = (unsigned short*)(ws + 0 * MB);
    // Region B (128..160MB): x_bf16, later reused as x1_bf16
    unsigned short* xbf  = (unsigned short*)(ws + 128 * MB);
    unsigned short* x1bf = xbf;
    // Region C (160..224MB): x1 fp32 (post-attn residual, LN1 in place)
    float* x1 = (float*)(ws + 160 * MB);
    // Region W (224..248MB): transposed bf16 weights
    unsigned short* Wqt = (unsigned short*)(ws + 224 * MB);
    unsigned short* Wkt = (unsigned short*)(ws + 226 * MB);
    unsigned short* Wvt = (unsigned short*)(ws + 228 * MB);
    unsigned short* Wot = (unsigned short*)(ws + 230 * MB);
    unsigned short* W1t = (unsigned short*)(ws + 232 * MB);
    unsigned short* W2t = (unsigned short*)(ws + 240 * MB);
    // Small buffers (248..250MB)
    unsigned short* KVb  = (unsigned short*)(ws + 248 * MB);
    float*          Ksum = (float*)(ws + 248 * MB + 512 * 1024);
    float*          zbuf = (float*)(ws + 249 * MB);

    // 1) convert x -> bf16
    cvt_f32_bf16_kernel<<<(N_TOK * D_MODEL / 4 + 255) / 256, 256, 0, stream>>>(
        x, xbf, N_TOK * D_MODEL / 4);

    // 2) transpose+convert weights: W[K][N] fp32 -> Wt[N][K] bf16
    transpose_cvt_kernel<<<dim3(D_MODEL / 32, D_MODEL / 32), 256, 0, stream>>>(Wq, Wqt, D_MODEL, D_MODEL);
    transpose_cvt_kernel<<<dim3(D_MODEL / 32, D_MODEL / 32), 256, 0, stream>>>(Wk, Wkt, D_MODEL, D_MODEL);
    transpose_cvt_kernel<<<dim3(D_MODEL / 32, D_MODEL / 32), 256, 0, stream>>>(Wv, Wvt, D_MODEL, D_MODEL);
    transpose_cvt_kernel<<<dim3(D_MODEL / 32, D_MODEL / 32), 256, 0, stream>>>(Wo, Wot, D_MODEL, D_MODEL);
    transpose_cvt_kernel<<<dim3(FFN / 32, D_MODEL / 32), 256, 0, stream>>>(W1, W1t, D_MODEL, FFN);
    transpose_cvt_kernel<<<dim3(D_MODEL / 32, FFN / 32), 256, 0, stream>>>(W2, W2t, FFN, D_MODEL);

    // 3) Q/K/V projections (elu+1 fused on Q,K)
    dim3 gQKV(D_MODEL / 128, N_TOK / 128);
    gemm_bf16_kernel<EPI_ELU1><<<gQKV, 256, 0, stream>>>(xbf, Wqt, N_TOK, D_MODEL, D_MODEL, bq, nullptr, Qf, nullptr);
    gemm_bf16_kernel<EPI_ELU1><<<gQKV, 256, 0, stream>>>(xbf, Wkt, N_TOK, D_MODEL, D_MODEL, bk, nullptr, Kf, nullptr);
    gemm_bf16_kernel<EPI_BIAS><<<gQKV, 256, 0, stream>>>(xbf, Wvt, N_TOK, D_MODEL, D_MODEL, bv, nullptr, Vb, nullptr);

    // 4) KV state + Ksum, then Z
    kv_state_kernel<<<NB * NH, 256, 0, stream>>>(Kf, Vb, KVb, Ksum);
    z_kernel<<<N_TOK, 256, 0, stream>>>(Qf, Ksum, zbuf);

    // 5) per-head attention output (WMMA, z fused)
    attn_out_kernel<<<dim3(SEQ / 128, NH, NB), 256, 0, stream>>>(Qf, KVb, zbuf, attn);

    // 6) O projection + residual (x) -> x1 fp32 ; LN1 in place + bf16 copy
    gemm_bf16_kernel<EPI_RESID><<<gQKV, 256, 0, stream>>>(attn, Wot, N_TOK, D_MODEL, D_MODEL, bo, x, nullptr, x1);
    layernorm_kernel<<<N_TOK, 256, 0, stream>>>(x1, gamma1, beta1, x1, x1bf);

    // 7) FFN: relu(x1@W1+b1) -> ffn1 bf16 ; ffn1@W2+b2 + x1 -> d_out fp32 ; LN2 in place
    gemm_bf16_kernel<EPI_RELU><<<dim3(FFN / 128, N_TOK / 128), 256, 0, stream>>>(
        x1bf, W1t, N_TOK, FFN, D_MODEL, b1, nullptr, ffn1, nullptr);
    gemm_bf16_kernel<EPI_RESID><<<gQKV, 256, 0, stream>>>(ffn1, W2t, N_TOK, D_MODEL, FFN, b2, x1, nullptr, outF);
    layernorm_kernel<<<N_TOK, 256, 0, stream>>>(outF, gamma2, beta2, outF, nullptr);
}
